// _SetAbstractionWithAttention_86165633892450
// MI455X (gfx1250) — matrix operand order repaired and checked by reading, not compiled, not run on hardware
//
#include <hip/hip_runtime.h>
#include <math.h>

// ---------------- problem constants (match reference) ----------------
#define BB      16
#define NN      4096
#define DFEAT   64
#define SS      1024      // NPOINT
#define KK      32        // NSAMPLE
#define INCH    67        // 3 + DFEAT
#define KP0     96        // INCH padded to multiple of 32 for WMMA
#define C_L0    64
#define C_L1    64
#define C_L2    128
#define MROWS   ((size_t)BB * SS * KK)   // 524288
#define EPS_BN  1e-5f

typedef __attribute__((ext_vector_type(16))) _Float16     v16h;
typedef __attribute__((ext_vector_type(8)))  _Float16     v8h;
typedef __attribute__((ext_vector_type(8)))  float        v8f;
typedef __attribute__((ext_vector_type(4)))  unsigned int u32x4;
typedef __attribute__((ext_vector_type(8)))  unsigned int u32x8;

// ---------------------------------------------------------------------
// small utility kernels
// ---------------------------------------------------------------------
__global__ void zero_f32_kernel(float* p, int n) {
    for (int i = blockIdx.x * blockDim.x + threadIdx.x; i < n;
         i += gridDim.x * blockDim.x)
        p[i] = 0.0f;
}

// convert W (oc x ic, f32) -> Wh (oc x kp, f16, zero padded)
__global__ void pack_weights_kernel(const float* __restrict__ W,
                                    _Float16* __restrict__ Wh,
                                    int oc, int ic, int kp) {
    int total = oc * kp;
    for (int e = blockIdx.x * blockDim.x + threadIdx.x; e < total;
         e += gridDim.x * blockDim.x) {
        int c = e % kp, o = e / kp;
        Wh[e] = (c < ic) ? (_Float16)W[o * ic + c] : (_Float16)0.0f;
    }
}

// ---------------------------------------------------------------------
// 1) Farthest point sampling: one block per batch, 1024 threads
//    writes new_xyz and fps_idx straight into d_out
// ---------------------------------------------------------------------
__global__ void fps_kernel(const float* __restrict__ xyz,
                           float* __restrict__ out_newxyz,
                           int* __restrict__ out_idx) {
    const int b = blockIdx.x, tid = threadIdx.x;
    __shared__ float sd[NN];
    __shared__ float rv[1024];
    __shared__ int   ri[1024];
    const float* base = xyz + (size_t)b * NN * 3;

    for (int i = tid; i < NN; i += 1024) sd[i] = 3.4e38f;
    __syncthreads();

    int far = NN / 2;
    for (int step = 0; step < SS; ++step) {
        if (tid == 0) {
            out_idx[b * SS + step] = far;
            size_t o = ((size_t)b * SS + step) * 3;
            out_newxyz[o + 0] = base[far * 3 + 0];
            out_newxyz[o + 1] = base[far * 3 + 1];
            out_newxyz[o + 2] = base[far * 3 + 2];
        }
        float cx = base[far * 3 + 0], cy = base[far * 3 + 1], cz = base[far * 3 + 2];
        float bv = -1.0f; int bi = 0;
        for (int i = tid; i < NN; i += 1024) {
            float dx = base[i * 3 + 0] - cx;
            float dy = base[i * 3 + 1] - cy;
            float dz = base[i * 3 + 2] - cz;
            float d  = dx * dx + dy * dy + dz * dz;
            float nd = fminf(sd[i], d);
            sd[i] = nd;
            if (nd > bv) { bv = nd; bi = i; }
        }
        rv[tid] = bv; ri[tid] = bi;
        __syncthreads();
        for (int off = 512; off > 0; off >>= 1) {
            if (tid < off && rv[tid + off] > rv[tid]) {
                rv[tid] = rv[tid + off]; ri[tid] = ri[tid + off];
            }
            __syncthreads();
        }
        far = ri[0];
        __syncthreads();
    }
}

// ---------------------------------------------------------------------
// 2) kNN: top-32 nearest of 4096 per query; grid (B, S/128), 128 thr
// ---------------------------------------------------------------------
__global__ void knn_kernel(const float* __restrict__ xyz,
                           const float* __restrict__ newxyz,
                           int* __restrict__ nbr) {
    const int b = blockIdx.x, tid = threadIdx.x;
    const int s = blockIdx.y * 128 + tid;
    const float* base = xyz + (size_t)b * NN * 3;
    size_t qo = ((size_t)b * SS + s) * 3;
    float qx = newxyz[qo + 0], qy = newxyz[qo + 1], qz = newxyz[qo + 2];

    __shared__ float kd[KK * 128];
    __shared__ int   ki[KK * 128];

    float worst = -1.0f; int wslot = 0;
    for (int j = 0; j < KK; ++j) {
        float dx = base[j * 3 + 0] - qx;
        float dy = base[j * 3 + 1] - qy;
        float dz = base[j * 3 + 2] - qz;
        float d = dx * dx + dy * dy + dz * dz;
        kd[j * 128 + tid] = d; ki[j * 128 + tid] = j;
        if (d > worst) { worst = d; wslot = j; }
    }
    for (int i = KK; i < NN; ++i) {
        float dx = base[i * 3 + 0] - qx;
        float dy = base[i * 3 + 1] - qy;
        float dz = base[i * 3 + 2] - qz;
        float d = dx * dx + dy * dy + dz * dz;
        if (d < worst) {
            kd[wslot * 128 + tid] = d; ki[wslot * 128 + tid] = i;
            worst = -1.0f;
            for (int j = 0; j < KK; ++j) {
                float v = kd[j * 128 + tid];
                if (v > worst) { worst = v; wslot = j; }
            }
        }
    }
    int* dst = nbr + ((size_t)b * SS + s) * KK;
    for (int j = 0; j < KK; ++j) dst[j] = ki[j * 128 + tid];
}

// ---------------------------------------------------------------------
// 3) Group: build f16 input matrix (MROWS x KP0): [rel_xyz | feat | 0pad]
// ---------------------------------------------------------------------
__global__ void group_kernel(const float* __restrict__ xyz,
                             const float* __restrict__ pts,
                             const float* __restrict__ newxyz,
                             const int* __restrict__ nbr,
                             _Float16* __restrict__ gmat) {
    const size_t total = MROWS * KP0;
    for (size_t e = (size_t)blockIdx.x * blockDim.x + threadIdx.x; e < total;
         e += (size_t)gridDim.x * blockDim.x) {
        int    c  = (int)(e % KP0);
        size_t m  = e / KP0;
        int    j  = (int)(m & (KK - 1));
        size_t bs = m >> 5;                 // b*SS + s
        int    b  = (int)(bs >> 10);
        int   ni  = nbr[bs * KK + j];
        float v = 0.0f;
        if (c < 3)
            v = xyz[((size_t)b * NN + ni) * 3 + c] - newxyz[bs * 3 + c];
        else if (c < INCH)
            v = pts[((size_t)b * NN + ni) * DFEAT + (c - 3)];
        gmat[e] = (_Float16)v;
    }
}

// ---------------------------------------------------------------------
// 4) GEMM: Y(M x nch, f32) = X(M x kp, f16) * Wh(nch x kp, f16)^T + bias
//
//    CDNA5 path: TDM (tensor_load_to_lds) stages a 64-row x KP A-tile
//    into LDS (one DMA per block, wave 0 issues it, TENSORcnt tracked),
//    waves read A fragments with ds_load_b128, B fragments (weights)
//    are register-resident across all 4 M-subtiles, accumulation via
//    v_wmma_f32_16x16x32_f16.
//
//    grid (M/64, nch/64), block 128 threads (4 waves); wave w owns the
//    16-column n-tile w and walks 4 x 16-row M-subtiles.
// ---------------------------------------------------------------------
template<int KTILES>
__global__ void gemm_wmma_kernel(const _Float16* __restrict__ X,
                                 const _Float16* __restrict__ Wh,
                                 const float* __restrict__ bias,
                                 float* __restrict__ Y, int nch) {
    constexpr int KP = KTILES * 32;
    __shared__ _Float16 atile[64 * KP];

    const int lane = threadIdx.x & 31;
    const int wave = threadIdx.x >> 5;
    const size_t m0 = (size_t)blockIdx.x * 64;
    const int n0 = blockIdx.y * 64 + wave * 16;
    const int hs = lane >> 4;            // 0 / 1 half-wave select
    const int colB = n0 + (lane & 15);

    // ---- wave 0: issue TDM load of the 64 x KP f16 A-tile into LDS ----
    if (wave == 0) {
        unsigned long long gaddr = (unsigned long long)(const void*)(X + m0 * KP);
        // generic LDS pointer low 32 bits == LDS byte offset (ISA aperture rule)
        unsigned int laddr = (unsigned int)(size_t)(const void*)atile;

        // D# group 0: count=1 | lds_addr | global_addr | type=2
        u32x4 g0;
        g0[0] = 1u;
        g0[1] = laddr;
        g0[2] = (unsigned int)gaddr;
        g0[3] = ((unsigned int)(gaddr >> 32) & 0x01FFFFFFu) | 0x80000000u;

        // D# group 1: data_size=2B; tensor_dim0=KP; tensor_dim1=MROWS;
        // tile_dim0=KP, tile_dim1=64; tensor_dim0_stride=KP (elements)
        u32x8 g1;
        g1[0] = (1u << 16);                                  // data_size code 1
        g1[1] = ((unsigned int)KP) << 16;                    // tensor_dim0[15:0]
        g1[2] = 0u;                                          // td0[31:16]=0, td1[15:0]=0
        g1[3] = (((unsigned int)KP) << 16) | 8u;             // tile_dim0 | td1[31:16] (=8 -> 524288)
        g1[4] = 64u;                                         // tile_dim1=64, tile_dim2=0
        g1[5] = (unsigned int)KP;                            // tensor_dim0_stride[31:0]
        g1[6] = 0u;
        g1[7] = 0u;

        asm volatile("tensor_load_to_lds %0, %1"
                     :: "s"(g0), "s"(g1)
                     : "memory");
        __builtin_amdgcn_s_wait_tensorcnt(0);
    }
    __syncthreads();

    // ---- preload B fragments (weights) for this wave's n-tile ----
    const _Float16* wp = Wh + (size_t)colB * KP + hs * 16;
    v16h breg[KTILES];
#pragma unroll
    for (int kt = 0; kt < KTILES; ++kt)
        breg[kt] = *(const v16h*)(wp + kt * 32);

    float bb = bias[colB];

    // ---- 4 M-subtiles, A fragments from LDS ----
#pragma unroll
    for (int msub = 0; msub < 4; ++msub) {
        v8f c;
#pragma unroll
        for (int i = 0; i < 8; ++i) c[i] = bb;

        const _Float16* ap = atile + (msub * 16 + (lane & 15)) * KP + hs * 16;
#pragma unroll
        for (int kt = 0; kt < KTILES; ++kt) {
            v8h alo = *(const v8h*)(ap + kt * 32);
            v8h ahi = *(const v8h*)(ap + kt * 32 + 8);
            v16h a;
#pragma unroll
            for (int i = 0; i < 8; ++i) { a[i] = alo[i]; a[8 + i] = ahi[i]; }
            c = __builtin_amdgcn_wmma_f32_16x16x32_f16(
                false, a, false, breg[kt], (short)0, c, false, false);
        }

        const size_t rbase = m0 + msub * 16 + hs * 8;
#pragma unroll
        for (int r = 0; r < 8; ++r)
            Y[(rbase + r) * nch + colB] = c[r];
    }
}

// ---------------------------------------------------------------------
// 5) Column stats (sum, sumsq) for BatchNorm (training-mode stats)
// ---------------------------------------------------------------------
__global__ void colstats_kernel(const float* __restrict__ Y, int rowsPerBlock,
                                int C, float* __restrict__ sum,
                                float* __restrict__ sq) {
    const int tid = threadIdx.x;          // 256 threads
    const int G = 256 / C;
    const int c = tid % C, g = tid / C;
    const size_t r0 = (size_t)blockIdx.x * rowsPerBlock;
    float s = 0.0f, s2 = 0.0f;
    for (int r = g; r < rowsPerBlock; r += G) {
        float v = Y[(r0 + r) * (size_t)C + c];
        s += v; s2 += v * v;
    }
    __shared__ float sh[256], sh2[256];
    sh[tid] = s; sh2[tid] = s2;
    __syncthreads();
    if (g == 0) {
        for (int gg = 1; gg < G; ++gg) { s += sh[gg * C + c]; s2 += sh2[gg * C + c]; }
        atomicAdd(&sum[c], s);
        atomicAdd(&sq[c], s2);
    }
}

__global__ void bn_finalize_kernel(const float* __restrict__ sum,
                                   const float* __restrict__ sq,
                                   const float* __restrict__ gamma,
                                   const float* __restrict__ beta,
                                   float invM, float* __restrict__ scale,
                                   float* __restrict__ shift) {
    int c = threadIdx.x;
    float mu = sum[c] * invM;
    float var = sq[c] * invM - mu * mu;
    float rs = rsqrtf(var + EPS_BN);
    float sc = gamma[c] * rs;
    scale[c] = sc;
    shift[c] = beta[c] - mu * sc;
}

// BN + ReLU + convert to f16 (input to next GEMM layer)
__global__ void bnrelu_f16_kernel(const float* __restrict__ Y,
                                  const float* __restrict__ scale,
                                  const float* __restrict__ shift,
                                  _Float16* __restrict__ X, size_t total, int C) {
    for (size_t e = (size_t)blockIdx.x * blockDim.x + threadIdx.x; e < total;
         e += (size_t)gridDim.x * blockDim.x) {
        int c = (int)(e % C);
        float v = Y[e] * scale[c] + shift[c];
        X[e] = (_Float16)fmaxf(v, 0.0f);
    }
}

// ---------------------------------------------------------------------
// 6) BN + ReLU + max over nsample -> pooled (B,S,128)
// ---------------------------------------------------------------------
__global__ void pool_kernel(const float* __restrict__ Y2,
                            const float* __restrict__ scale,
                            const float* __restrict__ shift,
                            float* __restrict__ pooled) {
    const size_t total = (size_t)BB * SS * C_L2;
    for (size_t e = (size_t)blockIdx.x * blockDim.x + threadIdx.x; e < total;
         e += (size_t)gridDim.x * blockDim.x) {
        int c = (int)(e % C_L2);
        size_t ms = e / C_L2;           // b*SS + s
        float sc = scale[c], sh = shift[c];
        float m = 0.0f;                  // relu outputs are >= 0
        for (int j = 0; j < KK; ++j) {
            float y = Y2[(ms * KK + j) * (size_t)C_L2 + c];
            m = fmaxf(m, fmaxf(y * sc + sh, 0.0f));
        }
        pooled[e] = m;
    }
}

// 7) avg + max over S per (b, c)
__global__ void reduceS_kernel(const float* __restrict__ pooled,
                               float* __restrict__ avg, float* __restrict__ mx) {
    const int b = blockIdx.x, c = threadIdx.x;     // 128 threads
    float s = 0.0f, m = -3.4e38f;
    for (int ss = 0; ss < SS; ++ss) {
        float v = pooled[((size_t)b * SS + ss) * C_L2 + c];
        s += v; m = fmaxf(m, v);
    }
    avg[b * C_L2 + c] = s * (1.0f / SS);
    mx[b * C_L2 + c] = m;
}

// 8) channel attention: sigmoid(aW2@relu(aW1@avg) + aW2@relu(aW1@mx))
__global__ void attn_kernel(const float* __restrict__ avg,
                            const float* __restrict__ mx,
                            const float* __restrict__ aW1,   // (R=16, 128)
                            const float* __restrict__ aW2,   // (128, R=16)
                            float* __restrict__ scl) {
    const int b = blockIdx.x, c = threadIdx.x;     // 128 threads
    __shared__ float sa[C_L2], sm[C_L2], hs[16];
    sa[c] = avg[b * C_L2 + c];
    sm[c] = mx[b * C_L2 + c];
    __syncthreads();
    if (c < 16) {
        float h1 = 0.0f, h2 = 0.0f;
        for (int cc = 0; cc < C_L2; ++cc) {
            float w = aW1[c * C_L2 + cc];
            h1 += w * sa[cc]; h2 += w * sm[cc];
        }
        hs[c] = fmaxf(h1, 0.0f) + fmaxf(h2, 0.0f);
    }
    __syncthreads();
    float acc = 0.0f;
    for (int r = 0; r < 16; ++r) acc += aW2[c * 16 + r] * hs[r];
    scl[b * C_L2 + c] = 1.0f / (1.0f + expf(-acc));
}

// 9) new_points = pooled * scale  -> d_out
__global__ void scaleout_kernel(const float* __restrict__ pooled,
                                const float* __restrict__ scl,
                                float* __restrict__ out) {
    const size_t total = (size_t)BB * SS * C_L2;
    for (size_t e = (size_t)blockIdx.x * blockDim.x + threadIdx.x; e < total;
         e += (size_t)gridDim.x * blockDim.x) {
        int c = (int)(e % C_L2);
        size_t ms = e / C_L2;
        int b = (int)(ms >> 10);
        out[e] = pooled[e] * scl[b * C_L2 + c];
    }
}

// ---------------------------------------------------------------------
// host-side launcher
// ---------------------------------------------------------------------
extern "C" void kernel_launch(void* const* d_in, const int* in_sizes, int n_in,
                              void* d_out, int out_size, void* d_ws, size_t ws_size,
                              hipStream_t stream) {
    const float* xyz = (const float*)d_in[0];
    const float* pts = (const float*)d_in[1];
    const float* W0  = (const float*)d_in[2];
    const float* b0  = (const float*)d_in[3];
    const float* g0  = (const float*)d_in[4];
    const float* be0 = (const float*)d_in[5];
    const float* W1  = (const float*)d_in[6];
    const float* b1  = (const float*)d_in[7];
    const float* g1  = (const float*)d_in[8];
    const float* be1 = (const float*)d_in[9];
    const float* W2  = (const float*)d_in[10];
    const float* b2  = (const float*)d_in[11];
    const float* g2  = (const float*)d_in[12];
    const float* be2 = (const float*)d_in[13];
    const float* aW1 = (const float*)d_in[14];
    const float* aW2 = (const float*)d_in[15];

    float* out_xyz = (float*)d_out;                               // (B,S,3)
    float* out_np  = out_xyz + (size_t)BB * SS * 3;               // (B,S,128)
    int*   out_idx = (int*)(out_np + (size_t)BB * SS * C_L2);     // (B,S) int32

    // scratch layout
    char* ws = (char*)d_ws;
    size_t off = 0;
    auto take = [&](size_t bytes) -> char* {
        char* p = ws + off;
        off = (off + bytes + 255) & ~(size_t)255;
        return p;
    };
    int*      nbr    = (int*)take(MROWS * sizeof(int));                   // (B,S,32)
    _Float16* bufA   = (_Float16*)take(MROWS * KP0 * sizeof(_Float16));   // f16 acts
    float*    bufB   = (float*)take(MROWS * C_L2 * sizeof(float));        // f32 layer out
    _Float16* W0h    = (_Float16*)take((size_t)C_L0 * KP0 * sizeof(_Float16));
    _Float16* W1h    = (_Float16*)take((size_t)C_L1 * 64 * sizeof(_Float16));
    _Float16* W2h    = (_Float16*)take((size_t)C_L2 * 64 * sizeof(_Float16));
    float*    stats  = (float*)take(256 * sizeof(float));                 // sum|sq
    float*    bnsc   = (float*)take(C_L2 * sizeof(float));
    float*    bnsh   = (float*)take(C_L2 * sizeof(float));
    float*    pooled = (float*)take((size_t)BB * SS * C_L2 * sizeof(float));
    float*    avgb   = (float*)take(BB * C_L2 * sizeof(float));
    float*    mxb    = (float*)take(BB * C_L2 * sizeof(float));
    float*    scl    = (float*)take(BB * C_L2 * sizeof(float));
    (void)in_sizes; (void)n_in; (void)out_size; (void)ws_size;

    const float invM = 1.0f / (float)MROWS;
    const int MB = (int)(MROWS / 64);   // 8192 blocks of 64 rows

    // weights -> padded f16
    pack_weights_kernel<<<24, 256, 0, stream>>>(W0, W0h, C_L0, INCH, KP0);
    pack_weights_kernel<<<16, 256, 0, stream>>>(W1, W1h, C_L1, 64, 64);
    pack_weights_kernel<<<32, 256, 0, stream>>>(W2, W2h, C_L2, 64, 64);

    // sampling / grouping
    fps_kernel<<<BB, 1024, 0, stream>>>(xyz, out_xyz, out_idx);
    knn_kernel<<<dim3(BB, SS / 128), 128, 0, stream>>>(xyz, out_xyz, nbr);
    group_kernel<<<4096, 256, 0, stream>>>(xyz, pts, out_xyz, nbr, bufA);

    // ---- layer 0: 96 -> 64 ----
    gemm_wmma_kernel<3><<<dim3(MB, 1), 128, 0, stream>>>(bufA, W0h, b0, bufB, C_L0);
    zero_f32_kernel<<<1, 256, 0, stream>>>(stats, 256);
    colstats_kernel<<<512, 256, 0, stream>>>(bufB, (int)(MROWS / 512), C_L0, stats, stats + 128);
    bn_finalize_kernel<<<1, C_L0, 0, stream>>>(stats, stats + 128, g0, be0, invM, bnsc, bnsh);
    bnrelu_f16_kernel<<<4096, 256, 0, stream>>>(bufB, bnsc, bnsh, bufA, MROWS * C_L0, C_L0);

    // ---- layer 1: 64 -> 64 ----
    gemm_wmma_kernel<2><<<dim3(MB, 1), 128, 0, stream>>>(bufA, W1h, b1, bufB, C_L1);
    zero_f32_kernel<<<1, 256, 0, stream>>>(stats, 256);
    colstats_kernel<<<512, 256, 0, stream>>>(bufB, (int)(MROWS / 512), C_L1, stats, stats + 128);
    bn_finalize_kernel<<<1, C_L1, 0, stream>>>(stats, stats + 128, g1, be1, invM, bnsc, bnsh);
    bnrelu_f16_kernel<<<4096, 256, 0, stream>>>(bufB, bnsc, bnsh, bufA, MROWS * C_L1, C_L1);

    // ---- layer 2: 64 -> 128 ----
    gemm_wmma_kernel<2><<<dim3(MB, 2), 128, 0, stream>>>(bufA, W2h, b2, bufB, C_L2);
    zero_f32_kernel<<<1, 256, 0, stream>>>(stats, 256);
    colstats_kernel<<<512, 256, 0, stream>>>(bufB, (int)(MROWS / 512), C_L2, stats, stats + 128);
    bn_finalize_kernel<<<1, C_L2, 0, stream>>>(stats, stats + 128, g2, be2, invM, bnsc, bnsh);

    // pooling + attention + output
    pool_kernel<<<2048, 256, 0, stream>>>(bufB, bnsc, bnsh, pooled);
    reduceS_kernel<<<BB, C_L2, 0, stream>>>(pooled, avgb, mxb);
    attn_kernel<<<BB, C_L2, 0, stream>>>(avgb, mxb, aW1, aW2, scl);
    scaleout_kernel<<<2048, 256, 0, stream>>>(pooled, scl, out_np);
}
